// LTAE2d_84877143703660
// MI455X (gfx1250) — matrix operand order repaired and checked by reading, not compiled, not run on hardware
//
#include <hip/hip_runtime.h>
#include <hip/hip_fp16.h>

typedef __attribute__((ext_vector_type(16))) _Float16 v16h;
typedef __attribute__((ext_vector_type(8)))  _Float16 v8h;
typedef __attribute__((ext_vector_type(8)))  float    v8f;

#define LN_EPS 1e-5f

// ---------------------------------------------------------------------------
// CDNA5 async global->LDS copies (ASYNCcnt-tracked, no VGPR round trip).
// LDS operand is the 32-bit LDS byte offset (low 32 bits of a generic
// shared-space pointer); address operand is a 64-bit global pointer.
// ---------------------------------------------------------------------------
__device__ __forceinline__ void async_g2l_b128(unsigned lds_off, const void* gptr) {
    asm volatile("global_load_async_to_lds_b128 %0, %1, off"
                 :: "v"(lds_off), "v"(gptr) : "memory");
}
__device__ __forceinline__ void async_g2l_b32(unsigned lds_off, const void* gptr) {
    asm volatile("global_load_async_to_lds_b32 %0, %1, off"
                 :: "v"(lds_off), "v"(gptr) : "memory");
}
__device__ __forceinline__ void wait_async0() {
    asm volatile("s_wait_asynccnt 0" ::: "memory");
}
__device__ __forceinline__ unsigned lds_off_of(const void* p) {
    return (unsigned)(size_t)p;
}

// ---------------------------------------------------------------------------
// WMMA helpers (CDNA5 wave32, V_WMMA_F32_16X16X32_F16)
// A fragment (16x32 f16) from an LDS row-major tile. Layout used here (same k order as load_b_frag):
//   lane grp=lane>>4, m=lane&15; K = grp*16 .. grp*16+15 (contiguous)
// ---------------------------------------------------------------------------
__device__ __forceinline__ v16h load_a_frag(const _Float16* base, int stride) {
    int lane = threadIdx.x & 31;
    int m    = lane & 15;
    int grp  = lane >> 4;
    const _Float16* p = base + m * stride + grp * 16;
    v8h lo = *(const v8h*)(p);
    v8h hi = *(const v8h*)(p + 8);
    return __builtin_shufflevector(lo, hi, 0,1,2,3,4,5,6,7,8,9,10,11,12,13,14,15);
}

// B fragment (32x16 f16) straight from a row-major [Nrows, K] f16 weight in
// global memory: per lane N = ncol0 + (lane&15), K = kbase + (lane>>4)*16 .. +15
// -> one aligned 32B contiguous load from the weight row (L2-resident).
__device__ __forceinline__ v16h load_b_frag(const _Float16* W, int K, int ncol0, int kbase) {
    int lane = threadIdx.x & 31;
    int n = ncol0 + (lane & 15);
    int k = kbase + ((lane >> 4) << 4);
    return *(const v16h*)(W + (size_t)n * K + k);
}

__device__ __forceinline__ v8f wmma_f16(v16h a, v16h b, v8f c) {
    return __builtin_amdgcn_wmma_f32_16x16x32_f16(false, a, false, b, (short)0, c, false, false);
}

// ---------------------------------------------------------------------------
// K0: convert weights to f16 scratch copies.
// Wc:[256,128]  Wqk:[128,256] (rows 0..63 = Wq, 64..127 = Wk)  Wm:[128,4096]
// ---------------------------------------------------------------------------
__global__ __launch_bounds__(256)
void k0_convert(const float* Wc, const float* Wq, const float* Wk, const float* Wm,
                _Float16* wc16, _Float16* wqk16, _Float16* wm16) {
    int gid = blockIdx.x * 256 + threadIdx.x;
    if (gid < 32768) {
        wc16[gid] = (_Float16)Wc[gid];
        int r = gid >> 8, c = gid & 255;
        float v = (r < 64) ? Wq[r * 256 + c] : Wk[(r - 64) * 256 + c];
        wqk16[gid] = (_Float16)v;
    }
    if (gid < 524288) wm16[gid] = (_Float16)Wm[gid];
}

// ---------------------------------------------------------------------------
// K1: gather + LayerNorm over C=128, emit A1 f16 [N*T, 128].
// One block = 32 consecutive w-pixels at one t -> coalesced 128B x-segments,
// gathered straight into LDS with async b32 loads.
// ---------------------------------------------------------------------------
__global__ __launch_bounds__(256)
void k1_ln1(const float* __restrict__ x, const float* __restrict__ g,
            const float* __restrict__ bvec, _Float16* __restrict__ A1) {
    const int T = 24, C = 128, HW = 4096;
    int blk  = blockIdx.x;
    int t    = blk % T;
    int pix0 = (blk / T) * 32;
    int bb   = pix0 >> 12;
    int rem  = pix0 & 4095;

    __shared__ float tile[128 * 32];   // [c][w]
    __shared__ float mS[32], rsS[32];

    int tid = threadIdx.x;
    int w   = tid & 31, cb = tid >> 5;
    const float* xb = x + (size_t)bb * C * T * HW + (size_t)t * HW + rem + w;
    unsigned tldc = lds_off_of(tile);
    for (int c = cb; c < C; c += 8)
        async_g2l_b32(tldc + (unsigned)(c * 32 + w) * 4, xb + (size_t)c * T * HW);
    wait_async0();
    __syncthreads();

    if (tid < 32) {
        float s = 0.f, sq = 0.f;
        for (int c = 0; c < C; c++) { float v = tile[c * 32 + tid]; s += v; sq += v * v; }
        float m = s * (1.0f / C);
        float var = sq * (1.0f / C) - m * m;
        mS[tid] = m; rsS[tid] = rsqrtf(var + LN_EPS);
    }
    __syncthreads();

    for (int i = 0; i < 16; i++) {
        int idx = tid + i * 256;
        int r = idx >> 7, c = idx & 127;
        float v = (tile[c * 32 + r] - mS[r]) * rsS[r] * g[c] + bvec[c];
        size_t row = (size_t)(pix0 + r) * T + t;
        A1[row * C + c] = (_Float16)v;
    }
}

// ---------------------------------------------------------------------------
// K2: OUT = LN2( A1 @ Wc^T + bc ), rows = N*T, K=128, Ncols=256 (f16 WMMA).
// M_BLK=32; 8 waves -> (2 m-tiles) x (4 ntile-groups of 4). Async A staging.
// ---------------------------------------------------------------------------
__global__ __launch_bounds__(256)
void k2_conv_ln2(const _Float16* __restrict__ A1, const _Float16* __restrict__ Wc16,
                 const float* __restrict__ bc, const float* __restrict__ g2,
                 const float* __restrict__ b2, _Float16* __restrict__ OUT) {
    const int K = 128, NC = 256, MB = 32;
    __shared__ _Float16 At[32 * 128];      // 8 KB
    __shared__ float    Dst[32 * 256];     // 32 KB
    __shared__ float    ps[256], pq[256];  // 2 KB partial sums
    __shared__ float    mS[32], rsS[32];

    int tid = threadIdx.x;
    size_t row0 = (size_t)blockIdx.x * MB;

    { // async-stage A tile: 512 x 16B
        unsigned adst = lds_off_of(At);
        const _Float16* src = A1 + row0 * K;
        for (int i = tid; i < 512; i += 256)
            async_g2l_b128(adst + (unsigned)i * 16, src + (size_t)i * 8);
    }
    wait_async0();
    __syncthreads();

    int wv = tid >> 5;
    int mt = wv & 1;
    int ng = wv >> 1;
    v8f acc[4] = {};
    for (int ks = 0; ks < K; ks += 32) {
        v16h a = load_a_frag(At + mt * 16 * K + ks, K);
        for (int nt = 0; nt < 4; nt++) {
            v16h bf = load_b_frag(Wc16, K, (ng * 4 + nt) * 16, ks);
            acc[nt] = wmma_f16(a, bf, acc[nt]);
        }
    }

    int lane = tid & 31, grp = lane >> 4, col = lane & 15;
    for (int nt = 0; nt < 4; nt++) {
        int nc = (ng * 4 + nt) * 16 + col;
        float bias = bc[nc];
        for (int r = 0; r < 8; r++)
            Dst[(mt * 16 + grp * 8 + r) * NC + nc] = acc[nt][r] + bias;
    }
    __syncthreads();

    { // parallel LN2 stats: 8 partial threads per row
        int r = tid >> 3, part = tid & 7;
        const float* row = Dst + r * NC + part * 32;
        float s = 0.f, sq = 0.f;
        for (int c = 0; c < 32; c++) { float v = row[c]; s += v; sq += v * v; }
        ps[tid] = s; pq[tid] = sq;
    }
    __syncthreads();
    if (tid < 32) {
        float S = 0.f, Q = 0.f;
        for (int k = 0; k < 8; k++) { S += ps[tid * 8 + k]; Q += pq[tid * 8 + k]; }
        float m = S * (1.0f / NC);
        float var = Q * (1.0f / NC) - m * m;
        mS[tid] = m; rsS[tid] = rsqrtf(var + LN_EPS);
    }
    __syncthreads();

    for (int i = 0; i < 32; i++) {
        int idx = tid + i * 256;
        int r = idx >> 8, c = idx & 255;
        float v = (Dst[r * NC + c] - mS[r]) * rsS[r] * g2[c] + b2[c];
        OUT[(row0 + r) * NC + c] = (_Float16)v;
    }
}

// ---------------------------------------------------------------------------
// K3: QK = OUT @ Wqk^T + [bq|bk]; rows = N*T, K=256, Ncols=128 (f16 WMMA).
// ---------------------------------------------------------------------------
__global__ __launch_bounds__(256)
void k3_qk(const _Float16* __restrict__ OUT, const _Float16* __restrict__ Wqk16,
           const float* __restrict__ bq, const float* __restrict__ bk,
           _Float16* __restrict__ QK) {
    const int K = 256, NC = 128, MB = 32;
    __shared__ _Float16 At[32 * 256];   // 16 KB
    int tid = threadIdx.x;
    size_t row0 = (size_t)blockIdx.x * MB;

    {
        unsigned adst = lds_off_of(At);
        const _Float16* src = OUT + row0 * K;
        for (int i = tid; i < 1024; i += 256)
            async_g2l_b128(adst + (unsigned)i * 16, src + (size_t)i * 8);
    }
    wait_async0();
    __syncthreads();

    int wv = tid >> 5;
    int mt = wv & 1;
    int ng = wv >> 1;     // 4 groups x 2 ntiles = 8 ntiles = 128 cols
    v8f acc[2] = {};
    for (int ks = 0; ks < K; ks += 32) {
        v16h a = load_a_frag(At + mt * 16 * K + ks, K);
        for (int nt = 0; nt < 2; nt++) {
            v16h bf = load_b_frag(Wqk16, K, (ng * 2 + nt) * 16, ks);
            acc[nt] = wmma_f16(a, bf, acc[nt]);
        }
    }

    int lane = tid & 31, grp = lane >> 4, col = lane & 15;
    for (int nt = 0; nt < 2; nt++) {
        int nc = (ng * 2 + nt) * 16 + col;
        float bias = (nc < 64) ? bq[nc] : bk[nc - 64];
        for (int r = 0; r < 8; r++) {
            size_t row = row0 + mt * 16 + grp * 8 + r;
            QK[row * NC + nc] = (_Float16)(acc[nt][r] + bias);
        }
    }
}

// ---------------------------------------------------------------------------
// K4: q mean-pool over T + BN1 + fc2 (64x64). Block = 16 pixels.
// ---------------------------------------------------------------------------
__global__ __launch_bounds__(256)
void k4_qpool(const _Float16* __restrict__ QK, const float* __restrict__ rm,
              const float* __restrict__ rv, const float* __restrict__ g,
              const float* __restrict__ bvec, const float* __restrict__ W2,
              const float* __restrict__ b2, float* __restrict__ q2) {
    const int T = 24;
    __shared__ float bnq[16 * 64];
    __shared__ float W2s[64 * 64];   // 16 KB
    int tid = threadIdx.x;
    size_t n0 = (size_t)blockIdx.x * 16;

    for (int i = tid; i < 4096; i += 256) W2s[i] = W2[i];
    for (int i = tid; i < 1024; i += 256) {
        int p = i >> 6, j = i & 63;
        size_t base = (n0 + p) * (size_t)T * 128 + j;
        float s = 0.f;
        for (int t = 0; t < T; t++) s += (float)QK[base + (size_t)t * 128];
        float qm = s * (1.0f / T);
        bnq[i] = (qm - rm[j]) * rsqrtf(rv[j] + LN_EPS) * g[j] + bvec[j];
    }
    __syncthreads();

    for (int i = tid; i < 1024; i += 256) {
        int p = i >> 6, j2 = i & 63;
        float s = b2[j2];
        const float* w = W2s + j2 * 64;
        const float* q = bnq + p * 64;
        for (int j = 0; j < 64; j++) s += q[j] * w[j];
        q2[(n0 + p) * 64 + j2] = s;
    }
}

// ---------------------------------------------------------------------------
// K5: attention scores (q.k / sqrt(4)), softmax over T, weighted sum over OUT
// -> O f16 [N, 4096]. Block = 2 pixels; OUT tile async-staged; weighted sum
// vectorized to v8h LDS reads (8 outputs / thread / step).
// ---------------------------------------------------------------------------
__global__ __launch_bounds__(256)
void k5_attn(const _Float16* __restrict__ QK, const _Float16* __restrict__ OUT,
             const float* __restrict__ q2, _Float16* __restrict__ O) {
    const int T = 24, DM = 256;
    __shared__ _Float16 Ot[2 * 24 * 256];   // 24 KB
    __shared__ float attn[2 * 16 * 24];     // 3 KB
    __shared__ float q2s[2 * 64];
    int tid = threadIdx.x;
    size_t n0 = (size_t)blockIdx.x * 2;

    {
        unsigned adst = lds_off_of(Ot);
        const _Float16* src = OUT + n0 * T * DM;
        for (int i = tid; i < 1536; i += 256)
            async_g2l_b128(adst + (unsigned)i * 16, src + (size_t)i * 8);
    }
    if (tid < 128) q2s[tid] = q2[n0 * 64 + tid];
    wait_async0();
    __syncthreads();

    for (int i = tid; i < 768; i += 256) {            // 2 px * 16 h * 24 t
        int p = i / 384, r = i % 384;
        int h = r / 24, t = r % 24;
        const _Float16* kp = QK + ((n0 + p) * T + t) * 128 + 64 + h * 4;
        const float* qp = q2s + p * 64 + h * 4;
        float s = 0.f;
        for (int d = 0; d < 4; d++) s += qp[d] * (float)kp[d];
        attn[i] = s * 0.5f;                            // / sqrt(DK=4)
    }
    __syncthreads();

    if (tid < 32) {                                    // softmax rows (p,h)
        float* a = attn + tid * 24;
        float mx = a[0];
        for (int t = 1; t < T; t++) mx = fmaxf(mx, a[t]);
        float s = 0.f;
        for (int t = 0; t < T; t++) { float e = __expf(a[t] - mx); a[t] = e; s += e; }
        float inv = 1.0f / s;
        for (int t = 0; t < T; t++) a[t] *= inv;
    }
    __syncthreads();

    // 2 px * 512 v8h chunks; chunk jc covers output cols jc*8 .. jc*8+7
    for (int i = tid; i < 1024; i += 256) {
        int p  = i >> 9;
        int jc = i & 511;
        int h  = jc >> 5;               // 32 chunks per head (256 cols)
        int d0 = (jc & 31) * 8;
        const float* a = attn + p * 384 + h * 24;
        const v8h* ot = (const v8h*)(Ot + p * T * DM + d0);
        float s[8] = {};
        for (int t = 0; t < T; t++) {
            v8h v = ot[t * (DM / 8)];
            float w = a[t];
            for (int k = 0; k < 8; k++) s[k] += w * (float)v[k];
        }
        v8h rvec;
        for (int k = 0; k < 8; k++) rvec[k] = (_Float16)s[k];
        *(v8h*)(O + (n0 + p) * 4096 + (size_t)jc * 8) = rvec;
    }
}

// ---------------------------------------------------------------------------
// K6: res = LN3( relu( BN2( O @ Wm^T + bm ) ) ); rows=N, K=4096, Ncols=128.
// M_BLK=16, 8 waves x 1 ntile each. Double-buffered async A staging overlaps
// the next chunk's global->LDS copy with this chunk's 8 WMMAs.
// ---------------------------------------------------------------------------
__global__ __launch_bounds__(256)
void k6_mlp(const _Float16* __restrict__ O, const _Float16* __restrict__ Wm16,
            const float* __restrict__ bm, const float* __restrict__ rm,
            const float* __restrict__ rv, const float* __restrict__ g2,
            const float* __restrict__ b2, const float* __restrict__ g3,
            const float* __restrict__ b3, float* __restrict__ out) {
    const int K = 4096, NC = 128, MB = 16, KC = 256, NCH = 16;
    __shared__ _Float16 At[2 * 16 * 256];  // 16 KB (double buffered)
    __shared__ float    Dst[16 * 128];     // 8 KB
    __shared__ float    mS[16], rsS[16];

    int tid = threadIdx.x;
    size_t n0 = (size_t)blockIdx.x * MB;
    int wv = tid >> 5;                  // ntile = wave id (8 waves -> 128 cols)
    unsigned at_lds = lds_off_of(At);
    v8f acc = {};

    { // prologue: stage chunk 0 into buffer 0
        const _Float16* src = O + n0 * K;
        for (int i = tid; i < 512; i += 256)
            async_g2l_b128(at_lds + (unsigned)i * 16,
                           src + ((size_t)(i >> 5) * K + (i & 31) * 8));
    }
    for (int ch = 0; ch < NCH; ch++) {
        wait_async0();
        __syncthreads();
        if (ch + 1 < NCH) { // stage next chunk into the other buffer
            const _Float16* src = O + n0 * K + (size_t)(ch + 1) * KC;
            unsigned dst = at_lds + (unsigned)(((ch + 1) & 1) * 8192);
            for (int i = tid; i < 512; i += 256)
                async_g2l_b128(dst + (unsigned)i * 16,
                               src + ((size_t)(i >> 5) * K + (i & 31) * 8));
            __builtin_prefetch(Wm16 + (size_t)(wv * 16) * K + (size_t)(ch + 1) * KC, 0, 0);
        }
        const _Float16* Ab = At + (ch & 1) * (16 * 256);
        for (int ks = 0; ks < KC; ks += 32) {
            v16h a  = load_a_frag(Ab + ks, KC);
            v16h bf = load_b_frag(Wm16, K, wv * 16, ch * KC + ks);
            acc = wmma_f16(a, bf, acc);
        }
    }
    __syncthreads();

    int lane = tid & 31, grp = lane >> 4, col = lane & 15;
    int nc = wv * 16 + col;
    for (int r = 0; r < 8; r++)
        Dst[(grp * 8 + r) * NC + nc] = acc[r] + bm[nc];
    __syncthreads();

    for (int i = tid; i < 2048; i += 256) {   // BN2 + ReLU in place
        int c = i & 127;
        float v = Dst[i];
        v = (v - rm[c]) * rsqrtf(rv[c] + LN_EPS) * g2[c] + b2[c];
        Dst[i] = fmaxf(v, 0.0f);
    }
    __syncthreads();

    if (tid < 16) {                            // LN3 row stats
        float s = 0.f, sq = 0.f;
        for (int c = 0; c < NC; c++) { float v = Dst[tid * NC + c]; s += v; sq += v * v; }
        float m = s * (1.0f / NC);
        float var = sq * (1.0f / NC) - m * m;
        mS[tid] = m; rsS[tid] = rsqrtf(var + LN_EPS);
    }
    __syncthreads();

    for (int i = tid; i < 2048; i += 256) {
        int c = i >> 4, r = i & 15;            // consecutive tid -> consecutive w
        size_t n = n0 + r;
        size_t bb = n >> 12;
        size_t hw = n & 4095;
        float v = (Dst[r * NC + c] - mS[r]) * rsS[r] * g3[c] + b3[c];
        out[((bb * 128 + (size_t)c) << 12) + hw] = v;
    }
}

// ---------------------------------------------------------------------------
extern "C" void kernel_launch(void* const* d_in, const int* in_sizes, int n_in,
                              void* d_out, int out_size, void* d_ws, size_t ws_size,
                              hipStream_t stream) {
    const float* x      = (const float*)d_in[0];
    const float* ln1_g  = (const float*)d_in[1];
    const float* ln1_b  = (const float*)d_in[2];
    const float* Wc     = (const float*)d_in[3];
    const float* bc     = (const float*)d_in[4];
    const float* ln2_g  = (const float*)d_in[5];
    const float* ln2_b  = (const float*)d_in[6];
    const float* Wq     = (const float*)d_in[7];
    const float* bq     = (const float*)d_in[8];
    const float* bn1_rm = (const float*)d_in[9];
    const float* bn1_rv = (const float*)d_in[10];
    const float* bn1_g  = (const float*)d_in[11];
    const float* bn1_b  = (const float*)d_in[12];
    const float* W2     = (const float*)d_in[13];
    const float* b2     = (const float*)d_in[14];
    const float* Wk     = (const float*)d_in[15];
    const float* bk     = (const float*)d_in[16];
    const float* Wm     = (const float*)d_in[17];
    const float* bm     = (const float*)d_in[18];
    const float* bn2_rm = (const float*)d_in[19];
    const float* bn2_rv = (const float*)d_in[20];
    const float* bn2_g  = (const float*)d_in[21];
    const float* bn2_b  = (const float*)d_in[22];
    const float* ln3_g  = (const float*)d_in[23];
    const float* ln3_b  = (const float*)d_in[24];
    float* out = (float*)d_out;

    const size_t N = 16384, T = 24, NT = N * T;

    // Workspace carving (256B aligned). Total ~542 MB.
    char* ws = (char*)d_ws;
    size_t off = 0;
    auto carve = [&](size_t bytes) {
        char* p = ws + off;
        off = (off + bytes + 255) & ~(size_t)255;
        return p;
    };
    _Float16* wc16  = (_Float16*)carve(32768 * 2);
    _Float16* wqk16 = (_Float16*)carve(32768 * 2);
    _Float16* wm16  = (_Float16*)carve(524288 * 2);
    _Float16* A1    = (_Float16*)carve(NT * 128 * 2);
    _Float16* OUTb  = (_Float16*)carve(NT * 256 * 2);
    _Float16* QK    = (_Float16*)carve(NT * 128 * 2);
    float*    q2    = (float*)   carve(N * 64 * 4);
    _Float16* Obuf  = (_Float16*)carve(N * 4096 * 2);
    (void)ws_size; (void)in_sizes; (void)n_in; (void)out_size;

    k0_convert<<<2048, 256, 0, stream>>>(Wc, Wq, Wk, Wm, wc16, wqk16, wm16);
    k1_ln1<<<(N / 32) * T, 256, 0, stream>>>(x, ln1_g, ln1_b, A1);
    k2_conv_ln2<<<NT / 32, 256, 0, stream>>>(A1, wc16, bc, ln2_g, ln2_b, OUTb);
    k3_qk<<<NT / 32, 256, 0, stream>>>(OUTb, wqk16, bq, bk, QK);
    k4_qpool<<<N / 16, 256, 0, stream>>>(QK, bn1_rm, bn1_rv, bn1_g, bn1_b, W2, b2, q2);
    k5_attn<<<N / 2, 256, 0, stream>>>(QK, OUTb, q2, Obuf);
    k6_mlp<<<N / 16, 256, 0, stream>>>(Obuf, wm16, bm, bn2_rm, bn2_rv, bn2_g, bn2_b,
                                       ln3_g, ln3_b, out);
}
